// Indexer_42537356100053
// MI455X (gfx1250) — compile-verified, matrix operand order checked by reading, not run on hardware
//
#include <hip/hip_runtime.h>
#include <hip/hip_bf16.h>
#include <math.h>

typedef __bf16 bf16_t;
typedef __attribute__((ext_vector_type(16))) __bf16 v16bf;
typedef __attribute__((ext_vector_type(8)))  __bf16 v8bf;
typedef __attribute__((ext_vector_type(8)))  float  v8f;
typedef __attribute__((ext_vector_type(4)))  int    v4i;

#define SEQ    4096
#define DIM    2048
#define RANK   1536
#define NH     16
#define HD     128
#define ROPE_D 64
#define TOPK_N 2048

#define AS1 __attribute__((address_space(1)))
#define AS3 __attribute__((address_space(3)))

#if __has_builtin(__builtin_amdgcn_global_load_async_to_lds_b128) && \
    __has_builtin(__builtin_amdgcn_s_wait_asynccnt)
#define USE_ASYNC_LDS 1
#else
#define USE_ASYNC_LDS 0
#endif

// ---------------------------------------------------------------------------
// bf16 A/B fragment load for v_wmma_f32_16x16x32_bf16.
// Per ISA 7.12.2 (16-bit A 16x32): lane L (0-15) holds row M=L, K = kbase+0..7
// in vector elems 0..7 and K = 16+kbase+0..7 in elems 8..15, where
// kbase = (lane>=16) ? 8 : 0.  B (32x16, column-per-lane) has the identical
// per-lane pattern when reading rows of a row-major K^T operand.
// `row` must already include the per-lane kbase offset.  Works for global
// (global_load_b128) and LDS (ds_load_b128) pointers alike.
// ---------------------------------------------------------------------------
__device__ __forceinline__ v16bf load_frag(const bf16_t* __restrict__ row, int k0) {
    v8bf lo = *reinterpret_cast<const v8bf*>(row + k0);
    v8bf hi = *reinterpret_cast<const v8bf*>(row + k0 + 16);
    return __builtin_shufflevector(lo, hi, 0,1,2,3,4,5,6,7,8,9,10,11,12,13,14,15);
}

// ---------------------------------------------------------------------------
// elementwise f32 -> bf16
// ---------------------------------------------------------------------------
__global__ void cvt_bf16_kernel(const float* __restrict__ src, bf16_t* __restrict__ dst, int n) {
    int i = blockIdx.x * blockDim.x + threadIdx.x;
    if (i < n) dst[i] = (bf16_t)src[i];
}

// ---------------------------------------------------------------------------
// C[M,N] = A[M,K] * B[N,K]^T  (both row-major over K).
// One wave per 16(M) x 64(N) strip: the A fragment is loaded once per k-chunk
// and reused across 4 N-subtiles (4x less A traffic than 16x16 tiling).
// ---------------------------------------------------------------------------
__global__ void wmma_gemm_bf16_kernel(const bf16_t* __restrict__ A, const bf16_t* __restrict__ B,
                                      float* __restrict__ C, int M, int N, int K) {
    const int lane = threadIdx.x & 31;
    const int wave = threadIdx.x >> 5;
    const int gid  = blockIdx.x * (blockDim.x >> 5) + wave;
    const int tn   = N >> 6;            // 64-column blocks
    const int mt   = gid / tn;
    const int nb   = gid % tn;
    if (mt * 16 >= M) return;

    const int r    = lane & 15;
    const int koff = (lane >> 4) * 8;   // kbase for A/B fragments
    const int hi8  = (lane >> 4) * 8;   // M offset for C/D rows

    const bf16_t* arow = A + (size_t)(mt * 16 + r) * (size_t)K + koff;
    const bf16_t* brow = B + (size_t)(nb * 64 + r) * (size_t)K + koff;

    v8f acc[4] = {};
    for (int k0 = 0; k0 < K; k0 += 32) {
        v16bf a = load_frag(arow, k0);
#pragma unroll
        for (int nt = 0; nt < 4; ++nt) {
            v16bf b = load_frag(brow + (size_t)(nt * 16) * (size_t)K, k0);
            acc[nt] = __builtin_amdgcn_wmma_f32_16x16x32_bf16(false, a, false, b,
                                                              (short)0, acc[nt], false, false);
        }
    }
    const int col = lane & 15;
#pragma unroll
    for (int nt = 0; nt < 4; ++nt)
#pragma unroll
        for (int e = 0; e < 8; ++e)
            C[(size_t)(mt * 16 + e + hi8) * (size_t)N + nb * 64 + nt * 16 + col] = acc[nt][e];
}

// ---------------------------------------------------------------------------
// q post: rope (first 64 dims) + FWHT(128) * 128^-0.5, f32 -> bf16
// one 128-thread block per (m,h) vector
// ---------------------------------------------------------------------------
__global__ void qpost_kernel(const float* __restrict__ qf, const float* __restrict__ freqs,
                             bf16_t* __restrict__ Qb) {
    __shared__ float v[HD];
    const int d   = threadIdx.x;
    const int vec = blockIdx.x;          // m*NH + h
    const int m   = vec >> 4;

    v[d] = qf[(size_t)vec * HD + d];
    __syncthreads();
    float val = v[d];
    if (d < ROPE_D) {
        int i = d >> 1;
        float fr = freqs[m * (ROPE_D / 2) + i];
        float c = cosf(fr), s = sinf(fr);
        float x1 = v[2 * i], x2 = v[2 * i + 1];
        val = (d & 1) ? (x1 * s + x2 * c) : (x1 * c - x2 * s);
    }
    __syncthreads();
    v[d] = val;
    __syncthreads();
#pragma unroll
    for (int hstep = 1; hstep < HD; hstep <<= 1) {
        float a = v[d], b = v[d ^ hstep];
        __syncthreads();
        v[d] = (d & hstep) ? (b - a) : (a + b);
        __syncthreads();
    }
    Qb[(size_t)vec * HD + d] = (bf16_t)(v[d] * 0.08838834764831845f); // 128^-0.5
}

// ---------------------------------------------------------------------------
// k post: layernorm(128) + rope + FWHT, f32 -> bf16. One block per row.
// ---------------------------------------------------------------------------
__global__ void kpost_kernel(const float* __restrict__ kl, const float* __restrict__ freqs,
                             const float* __restrict__ g, const float* __restrict__ bb,
                             bf16_t* __restrict__ Kb) {
    __shared__ float v[HD];
    __shared__ float red[HD];
    const int d = threadIdx.x;
    const int m = blockIdx.x;

    float x = kl[(size_t)m * HD + d];
    red[d] = x;
    __syncthreads();
    for (int s = 64; s > 0; s >>= 1) { if (d < s) red[d] += red[d + s]; __syncthreads(); }
    float mu = red[0] * (1.0f / HD);
    __syncthreads();
    float dev = x - mu;
    red[d] = dev * dev;
    __syncthreads();
    for (int s = 64; s > 0; s >>= 1) { if (d < s) red[d] += red[d + s]; __syncthreads(); }
    float var = red[0] * (1.0f / HD);
    float y = dev * rsqrtf(var + 1e-5f) * g[d] + bb[d];
    __syncthreads();
    v[d] = y;
    __syncthreads();

    float val = v[d];
    if (d < ROPE_D) {
        int i = d >> 1;
        float fr = freqs[m * (ROPE_D / 2) + i];
        float c = cosf(fr), s = sinf(fr);
        float x1 = v[2 * i], x2 = v[2 * i + 1];
        val = (d & 1) ? (x1 * s + x2 * c) : (x1 * c - x2 * s);
    }
    __syncthreads();
    v[d] = val;
    __syncthreads();
#pragma unroll
    for (int hstep = 1; hstep < HD; hstep <<= 1) {
        float a = v[d], b = v[d ^ hstep];
        __syncthreads();
        v[d] = (d & hstep) ? (b - a) : (a + b);
        __syncthreads();
    }
    Kb[(size_t)m * HD + d] = (bf16_t)(v[d] * 0.08838834764831845f);
}

// ---------------------------------------------------------------------------
// w[m,h] = (x[m,:] . wp[h,:]) * (H*D)^-0.5   — one block per row m
// ---------------------------------------------------------------------------
__global__ void wproj_kernel(const float* __restrict__ x, const float* __restrict__ wp,
                             float* __restrict__ w) {
    __shared__ float xs[DIM];
    const int m = blockIdx.x;
    for (int t = threadIdx.x; t < DIM; t += blockDim.x) xs[t] = x[(size_t)m * DIM + t];
    __syncthreads();
    const int lane = threadIdx.x & 31;
    const int wave = threadIdx.x >> 5;
    for (int h = wave; h < NH; h += 8) {
        float s = 0.f;
        for (int k = lane; k < DIM; k += 32) s += xs[k] * wp[h * DIM + k];
#pragma unroll
        for (int off = 16; off > 0; off >>= 1) s += __shfl_xor(s, off, 32);
        if (lane == 0) w[m * NH + h] = s * 0.02209708691207961f; // 1/sqrt(2048)
    }
}

// ---------------------------------------------------------------------------
// Fused score. Block = 8 waves covering 128(M) x 64(N); all waves share one
// 64-row K tile staged in LDS (async-to-LDS when available), so per-head B
// fragments come from ds_load_b128 instead of redundant global traffic.
// per head: logits via 4 k-chunks x 4 n-subtiles of wmma; relu * w; mask.
// ---------------------------------------------------------------------------
__global__ void score_kernel(const bf16_t* __restrict__ Q, const bf16_t* __restrict__ Kb,
                             const float* __restrict__ W, float* __restrict__ S) {
    __shared__ bf16_t ktile[64 * HD];    // 16 KB, shared by all 8 waves
    __shared__ float  wlds[8][256];      // per-wave w[m0..m0+15][0..15]

    const int lane = threadIdx.x & 31;
    const int wave = threadIdx.x >> 5;
    const int mb   = blockIdx.x >> 6;    // 32 row blocks of 128
    const int nb   = blockIdx.x & 63;    // 64 col blocks of 64
    const int m0   = mb * 128 + wave * 16;
    const int n0   = nb * 64;
    const int r    = lane & 15;
    const int koff = (lane >> 4) * 8;
    const int hi8  = (lane >> 4) * 8;

    // ---- stage K tile (64 rows x 128 bf16 = 1024 x 16B chunks) ----
    const int tid = threadIdx.x;
#if USE_ASYNC_LDS
    {
        const bf16_t* gsrc = Kb + (size_t)n0 * HD;
#pragma unroll
        for (int i = 0; i < 4; ++i) {
            int idx = i * 256 + tid;     // 16-byte chunk id
            __builtin_amdgcn_global_load_async_to_lds_b128(
                (AS1 v4i*)(gsrc + idx * 8),
                (AS3 v4i*)(&ktile[idx * 8]), 0, 0);
        }
        __builtin_amdgcn_s_wait_asynccnt(0);
    }
#else
    for (int i = tid; i < 64 * HD / 8; i += 256)
        *reinterpret_cast<v8bf*>(&ktile[i * 8]) =
            *reinterpret_cast<const v8bf*>(Kb + (size_t)n0 * HD + i * 8);
#endif
    // ---- stage w for this wave's 16 rows ----
    for (int t = lane; t < 256; t += 32)
        wlds[wave][t] = W[(m0 + (t >> 4)) * NH + (t & 15)];
    __syncthreads();

    const bf16_t* arowbase = Q + (size_t)(m0 + r) * (size_t)DIM + koff;
    const bf16_t* browl    = &ktile[r * HD + koff];       // LDS B fragments

    v8f sc[4] = {};
    for (int h = 0; h < NH; ++h) {
        v8f lg[4] = {};
        const bf16_t* arow = arowbase + h * HD;
#pragma unroll
        for (int kc = 0; kc < 4; ++kc) {
            v16bf a = load_frag(arow, kc * 32);
#pragma unroll
            for (int nt = 0; nt < 4; ++nt) {
                v16bf b = load_frag(browl + nt * 16 * HD, kc * 32);
                lg[nt] = __builtin_amdgcn_wmma_f32_16x16x32_bf16(false, a, false, b,
                                                                 (short)0, lg[nt], false, false);
            }
        }
#pragma unroll
        for (int nt = 0; nt < 4; ++nt)
#pragma unroll
            for (int e = 0; e < 8; ++e)
                sc[nt][e] += fmaxf(lg[nt][e], 0.f) * wlds[wave][(e + hi8) * 16 + h];
    }

    const int col = lane & 15;
#pragma unroll
    for (int nt = 0; nt < 4; ++nt) {
#pragma unroll
        for (int e = 0; e < 8; ++e) {
            int mrow = m0 + e + hi8;
            int ncol = n0 + nt * 16 + col;
            float val = sc[nt][e] + ((ncol <= mrow) ? 0.f : -1e9f);
            S[(size_t)mrow * SEQ + ncol] = val;
        }
    }
}

// ---------------------------------------------------------------------------
// Per-row top-k via full bitonic sort in LDS (descending value, ascending
// index tie-break == jax.lax.top_k stability). 512 threads, 32KB LDS.
// ---------------------------------------------------------------------------
__global__ void topk_kernel(const float* __restrict__ S, int* __restrict__ out) {
    __shared__ float sval[SEQ];
    __shared__ int   sidx[SEQ];
    const int m = blockIdx.x;
    for (int t = threadIdx.x; t < SEQ; t += blockDim.x) {
        sval[t] = S[(size_t)m * SEQ + t];
        sidx[t] = t;
    }
    __syncthreads();
    for (int k = 2; k <= SEQ; k <<= 1) {
        for (int j = k >> 1; j > 0; j >>= 1) {
            for (int t = threadIdx.x; t < SEQ; t += blockDim.x) {
                int l = t ^ j;
                if (l > t) {
                    float va = sval[t], vb = sval[l];
                    int   ia = sidx[t], ib = sidx[l];
                    bool aBefore = (va > vb) || (va == vb && ia < ib);
                    bool desc    = ((t & k) == 0);
                    if (aBefore != desc) {
                        sval[t] = vb; sval[l] = va;
                        sidx[t] = ib; sidx[l] = ia;
                    }
                }
            }
            __syncthreads();
        }
    }
    for (int t = threadIdx.x; t < TOPK_N; t += blockDim.x)
        out[(size_t)m * TOPK_N + t] = sidx[t];
}

// ---------------------------------------------------------------------------
extern "C" void kernel_launch(void* const* d_in, const int* in_sizes, int n_in,
                              void* d_out, int out_size, void* d_ws, size_t ws_size,
                              hipStream_t stream) {
    (void)in_sizes; (void)n_in; (void)out_size; (void)ws_size;
    const float* x   = (const float*)d_in[0];   // (1,4096,2048)
    const float* qr  = (const float*)d_in[1];   // (1,4096,1536)
    const float* fr  = (const float*)d_in[2];   // (4096,32)
    /* d_in[3] mask computed analytically */
    const float* wqb = (const float*)d_in[4];   // (2048,1536)
    const float* wk  = (const float*)d_in[5];   // (128,2048)
    const float* kg  = (const float*)d_in[6];   // (128,)
    const float* kb  = (const float*)d_in[7];   // (128,)
    const float* wp  = (const float*)d_in[8];   // (16,2048)
    int* out = (int*)d_out;                     // (1,4096,2048) int32

    char* ws = (char*)d_ws;
    size_t off = 0;
    auto alloc = [&](size_t bytes) {
        void* p = ws + off;
        off = (off + bytes + 255) & ~(size_t)255;
        return p;
    };
    bf16_t* qr_b = (bf16_t*)alloc((size_t)SEQ * RANK * 2);
    bf16_t* wq_b = (bf16_t*)alloc((size_t)NH * HD * RANK * 2);
    bf16_t* x_b  = (bf16_t*)alloc((size_t)SEQ * DIM * 2);
    bf16_t* wk_b = (bf16_t*)alloc((size_t)HD * DIM * 2);
    float*  qf   = (float*)alloc((size_t)SEQ * DIM * 4);
    float*  kl   = (float*)alloc((size_t)SEQ * HD * 4);
    float*  wbuf = (float*)alloc((size_t)SEQ * NH * 4);
    bf16_t* Qb   = (bf16_t*)alloc((size_t)SEQ * DIM * 2);
    bf16_t* Kb   = (bf16_t*)alloc((size_t)SEQ * HD * 2);
    float*  sc   = (float*)alloc((size_t)SEQ * SEQ * 4);

    // 1) convert operands to bf16
    {
        int n;
        n = SEQ * RANK;     cvt_bf16_kernel<<<(n + 255) / 256, 256, 0, stream>>>(qr,  qr_b, n);
        n = NH * HD * RANK; cvt_bf16_kernel<<<(n + 255) / 256, 256, 0, stream>>>(wqb, wq_b, n);
        n = SEQ * DIM;      cvt_bf16_kernel<<<(n + 255) / 256, 256, 0, stream>>>(x,   x_b,  n);
        n = HD * DIM;       cvt_bf16_kernel<<<(n + 255) / 256, 256, 0, stream>>>(wk,  wk_b, n);
    }
    // 2) q = qr @ wq_b^T  (4096 x 2048 x 1536), 16x64 strips per wave
    wmma_gemm_bf16_kernel<<<(SEQ / 16) * (DIM / 64) / 8, 256, 0, stream>>>(
        qr_b, wq_b, qf, SEQ, DIM, RANK);
    // 3) k_lin = x @ wk^T  (4096 x 128 x 2048)
    wmma_gemm_bf16_kernel<<<(SEQ / 16) * (HD / 64) / 8, 256, 0, stream>>>(
        x_b, wk_b, kl, SEQ, HD, DIM);
    // 4) w projection
    wproj_kernel<<<SEQ, 256, 0, stream>>>(x, wp, wbuf);
    // 5) post-process q and k (rope + fwht [+ layernorm for k])
    qpost_kernel<<<SEQ * NH, HD, 0, stream>>>(qf, fr, Qb);
    kpost_kernel<<<SEQ, HD, 0, stream>>>(kl, fr, kg, kb, Kb);
    // 6) fused relu-weighted score + causal mask (128x64 per block)
    score_kernel<<<(SEQ / 128) * (SEQ / 64), 256, 0, stream>>>(Qb, Kb, wbuf, sc);
    // 7) per-row top-k indices
    topk_kernel<<<SEQ, 512, 0, stream>>>(sc, out);
}
